// CoAttention_28338194219585
// MI455X (gfx1250) — matrix operand order repaired and checked by reading, not compiled, not run on hardware
//
#include <hip/hip_runtime.h>

#define LDIM 2048
#define BDIM 8
#define DDIM 768
#define NROWS (BDIM * LDIM)   // 16384 = rows of y / out
#define EPS_BN 1e-5f
#define EPS_LN 1e-5f

typedef __attribute__((ext_vector_type(16))) __bf16 bf16x16;
typedef __attribute__((ext_vector_type(8)))  float  floatx8;
typedef __attribute__((ext_vector_type(4)))  unsigned int u32x4;

typedef int v4i __attribute__((vector_size(16)));
typedef __attribute__((address_space(1))) v4i gv4i;   // global 128-bit chunk
typedef __attribute__((address_space(3))) v4i lv4i;   // LDS 128-bit chunk

union BfVec {
  bf16x16 v;
  u32x4 q[2];
  unsigned int u[8];
};

__device__ __forceinline__ unsigned short f2bf_bits(float f) {
  unsigned int x = __float_as_uint(f);
  x += 0x7FFFu + ((x >> 16) & 1u);           // round-to-nearest-even
  return (unsigned short)(x >> 16);
}

__device__ __forceinline__ floatx8 wmma_bf16(bf16x16 a, bf16x16 b, floatx8 c) {
  return __builtin_amdgcn_wmma_f32_16x16x32_bf16(false, a, false, b, (short)0, c,
                                                 false, false);
}

// gfx1250 async copy: 16B global -> LDS, tracked by ASYNCcnt
__device__ __forceinline__ void async_cp_b128(const unsigned short* g,
                                              unsigned short* l) {
  __builtin_amdgcn_global_load_async_to_lds_b128((gv4i*)g, (lv4i*)l, 0, 0);
}

// ---------------- casts ----------------

__global__ void k_cast_q(const float* __restrict__ x, unsigned short* __restrict__ q) {
  size_t idx = (size_t)blockIdx.x * blockDim.x + threadIdx.x;
  int d = (int)(idx % DDIM);
  int b = (int)((idx / DDIM) % BDIM);
  int l = (int)(idx / ((size_t)DDIM * BDIM));
  q[((size_t)b * LDIM + l) * DDIM + d] = f2bf_bits(x[idx]);
}

__global__ void k_cast_kv(const float* __restrict__ xsrc,
                          unsigned short* __restrict__ kbf,
                          unsigned short* __restrict__ ktbf) {
  size_t idx = (size_t)blockIdx.x * blockDim.x + threadIdx.x;
  int d = (int)(idx % DDIM);
  int b = (int)((idx / DDIM) % BDIM);
  int l = (int)(idx / ((size_t)DDIM * BDIM));
  unsigned short v = f2bf_bits(xsrc[idx]);
  kbf [((size_t)b * LDIM + l) * DDIM + d] = v;
  ktbf[((size_t)b * DDIM + d) * LDIM + l] = v;
}

__global__ void k_cast_w(const float* __restrict__ W, unsigned short* __restrict__ wbf) {
  size_t idx = (size_t)blockIdx.x * blockDim.x + threadIdx.x;
  wbf[idx] = f2bf_bits(W[idx]);
}

// ---------------- unified LDS-tiled bf16 GEMM ----------------
// C[M x NTOT] = A[M x KDIM] * B^T (Bm is [NTOT x KDIM], i.e. col-major B)
// 256 threads, block tile 128x128, 8 waves as 4(M) x 2(N), wave tile 32x64.
// A/B K-panels (128x32 bf16) double-buffered in LDS via async copies.
// OUT: 0 = f32 * scale, 1 = bf16, 2 = f32 + bias[col]
#define LPAD 40   // LDS row pitch in bf16 elements (80B: conflict-free b128 reads)

template <int KDIM, int NTOT, int OUT>
__global__ void __launch_bounds__(256) k_gemm(
    const unsigned short* __restrict__ A, const unsigned short* __restrict__ Bm,
    void* __restrict__ outp, const float* __restrict__ bias, float scale) {
  constexpr int NBLK = NTOT / 128;
  constexpr int NSTEP = KDIM / 32;
  const int tid = threadIdx.x;
  const int mbase = (blockIdx.x / NBLK) * 128;
  const int nbase = (blockIdx.x % NBLK) * 128;

  __shared__ unsigned short sA[2][128 * LPAD];
  __shared__ unsigned short sB[2][128 * LPAD];

  // staging assignment: 512 chunks of 16B per panel, 2 per thread
  const int ar0 = tid >> 2, ac0 = (tid & 3) * 8;
  const int ar1 = (tid + 256) >> 2, ac1 = ((tid + 256) & 3) * 8;

  const unsigned short* Arow0 = A + (size_t)(mbase + ar0) * KDIM + ac0;
  const unsigned short* Arow1 = A + (size_t)(mbase + ar1) * KDIM + ac1;
  const unsigned short* Brow0 = Bm + (size_t)(nbase + ar0) * KDIM + ac0;
  const unsigned short* Brow1 = Bm + (size_t)(nbase + ar1) * KDIM + ac1;

  auto stage = [&](int bi, int kk) {
    async_cp_b128(Arow0 + kk, &sA[bi][ar0 * LPAD + ac0]);
    async_cp_b128(Arow1 + kk, &sA[bi][ar1 * LPAD + ac1]);
    async_cp_b128(Brow0 + kk, &sB[bi][ar0 * LPAD + ac0]);
    async_cp_b128(Brow1 + kk, &sB[bi][ar1 * LPAD + ac1]);
  };

  const int lane = tid & 31;
  const int wid  = tid >> 5;
  const int wm = wid >> 1, wn = wid & 1;    // 4 x 2 wave grid
  const int lrow = lane & 15, khalf = lane >> 4;

  floatx8 acc[2][4];
#pragma unroll
  for (int i = 0; i < 2; ++i)
#pragma unroll
    for (int j = 0; j < 4; ++j) acc[i][j] = (floatx8)0.0f;

  stage(0, 0);

  for (int s = 0; s < NSTEP; ++s) {
    if (s + 1 < NSTEP) {
      stage((s + 1) & 1, (s + 1) * 32);
      asm volatile("s_wait_asynccnt 0x4" ::: "memory");   // prev panel landed
    } else {
      asm volatile("s_wait_asynccnt 0x0" ::: "memory");
    }
    __syncthreads();

    const unsigned short* pa = sA[s & 1];
    const unsigned short* pb = sB[s & 1];

    BfVec a0, a1;
    {
      const unsigned short* q0 = pa + (wm * 32 + lrow) * LPAD + khalf * 8;
      a0.q[0] = *(const u32x4*)q0;
      a0.q[1] = *(const u32x4*)(q0 + 16);
      const unsigned short* q1 = q0 + 16 * LPAD;
      a1.q[0] = *(const u32x4*)q1;
      a1.q[1] = *(const u32x4*)(q1 + 16);
    }
#pragma unroll
    for (int n = 0; n < 4; ++n) {
      const unsigned short* qb = pb + (wn * 64 + n * 16 + lrow) * LPAD + khalf * 8;
      BfVec bb;
      bb.q[0] = *(const u32x4*)qb;
      bb.q[1] = *(const u32x4*)(qb + 16);
      acc[0][n] = wmma_bf16(a0.v, bb.v, acc[0][n]);
      acc[1][n] = wmma_bf16(a1.v, bb.v, acc[1][n]);
    }
    __syncthreads();
  }

#pragma unroll
  for (int mi = 0; mi < 2; ++mi) {
    int rowbase = mbase + wm * 32 + mi * 16 + khalf * 8;
#pragma unroll
    for (int n = 0; n < 4; ++n) {
      int col = nbase + wn * 64 + n * 16 + lrow;
      float bv = (OUT == 2) ? bias[col] : 0.0f;
#pragma unroll
      for (int r = 0; r < 8; ++r) {
        float v = acc[mi][n][r];
        if (OUT == 0)
          ((float*)outp)[(size_t)(rowbase + r) * NTOT + col] = v * scale;
        else if (OUT == 1)
          ((unsigned short*)outp)[(size_t)(rowbase + r) * NTOT + col] = f2bf_bits(v);
        else
          ((float*)outp)[(size_t)(rowbase + r) * NTOT + col] = v + bv;
      }
    }
  }
}

// ---------------- row max + 1/sum(exp) (per b) ----------------

__global__ void k_rowstats(const float* __restrict__ S, float* __restrict__ stats) {
  const int lane = threadIdx.x & 31;
  const int row  = blockIdx.x * (blockDim.x >> 5) + (threadIdx.x >> 5);
  const float* p = S + (size_t)row * LDIM + lane;
  float m = -3.0e38f, s = 0.0f;
  for (int i = 0; i < LDIM / 32; ++i) {
    float v = p[i * 32];
    float nm = fmaxf(m, v);
    s = s * __expf(m - nm) + __expf(v - nm);
    m = nm;
  }
  for (int off = 16; off > 0; off >>= 1) {
    float m2 = __shfl_xor(m, off);
    float s2 = __shfl_xor(s, off);
    float nm = fmaxf(m, m2);
    s = s * __expf(m - nm) + s2 * __expf(m2 - nm);
    m = nm;
  }
  if (lane == 0) {
    stats[row * 2]     = m;
    stats[row * 2 + 1] = 1.0f / s;
  }
}

// ---------------- P = softmax(S) in bf16 (per b) ----------------

__global__ void k_softmax(const float* __restrict__ S, const float* __restrict__ stats,
                          unsigned short* __restrict__ P) {
  size_t idx = (size_t)blockIdx.x * blockDim.x + threadIdx.x;
  int row = (int)(idx >> 11);   // / LDIM
  float mx = stats[row * 2], is = stats[row * 2 + 1];
  P[idx] = f2bf_bits(__expf(S[idx] - mx) * is);
}

// ---------------- BN stats / finalize / apply ----------------

__global__ void k_zero(float* __restrict__ p) {
  p[blockIdx.x * blockDim.x + threadIdx.x] = 0.0f;
}

__global__ void k_bnacc(const float* __restrict__ y, float* __restrict__ sums) {
  const int tid = threadIdx.x;
  const int base = blockIdx.x * 64;
  float s[3] = {0.f, 0.f, 0.f}, q[3] = {0.f, 0.f, 0.f};
  for (int r = 0; r < 64; ++r) {
    const float* row = y + (size_t)(base + r) * DDIM;
#pragma unroll
    for (int j = 0; j < 3; ++j) {
      float v = row[tid + j * 256];
      s[j] += v;
      q[j] += v * v;
    }
  }
#pragma unroll
  for (int j = 0; j < 3; ++j) {
    atomicAdd(&sums[tid + j * 256], s[j]);
    atomicAdd(&sums[DDIM + tid + j * 256], q[j]);
  }
}

__global__ void k_bnfin(const float* __restrict__ sums, const float* __restrict__ g,
                        const float* __restrict__ beta, float* __restrict__ ss) {
  int e = blockIdx.x * blockDim.x + threadIdx.x;
  const float inv_n = 1.0f / (float)NROWS;
  float mean = sums[e] * inv_n;
  float var  = sums[DDIM + e] * inv_n - mean * mean;
  float sc = g[e] * rsqrtf(var + EPS_BN);
  ss[e] = sc;
  ss[DDIM + e] = beta[e] - mean * sc;
}

__global__ void k_apply(const float* __restrict__ y, const float* __restrict__ ss,
                        const float* __restrict__ alpha, float* __restrict__ out) {
  size_t idx = (size_t)blockIdx.x * blockDim.x + threadIdx.x;
  int e = (int)(idx % DDIM);
  int l = (int)((idx / DDIM) % LDIM);
  int b = (int)(idx / ((size_t)LDIM * DDIM));
  float z = y[idx] * ss[e] + ss[DDIM + e];
  float a = alpha[0];
  z = z > 0.0f ? z : a * z;
  out[((size_t)l * BDIM + b) * DDIM + e] += z;
}

// ---------------- final LayerNorm (in place on d_out) ----------------

__global__ void k_ln(float* __restrict__ out, const float* __restrict__ g,
                     const float* __restrict__ beta) {
  __shared__ float ssum[256];
  __shared__ float ssq[256];
  const int tid = threadIdx.x;
  float* p = out + (size_t)blockIdx.x * DDIM;
  float v0 = p[tid], v1 = p[tid + 256], v2 = p[tid + 512];
  ssum[tid] = v0 + v1 + v2;
  ssq[tid]  = v0 * v0 + v1 * v1 + v2 * v2;
  __syncthreads();
  for (int off = 128; off > 0; off >>= 1) {
    if (tid < off) {
      ssum[tid] += ssum[tid + off];
      ssq[tid]  += ssq[tid + off];
    }
    __syncthreads();
  }
  const float inv_d = 1.0f / (float)DDIM;
  float mu  = ssum[0] * inv_d;
  float var = ssq[0] * inv_d - mu * mu;
  float rs  = rsqrtf(var + EPS_LN);
  p[tid]       = (v0 - mu) * rs * g[tid]       + beta[tid];
  p[tid + 256] = (v1 - mu) * rs * g[tid + 256] + beta[tid + 256];
  p[tid + 512] = (v2 - mu) * rs * g[tid + 512] + beta[tid + 512];
}

// ---------------- launch ----------------

extern "C" void kernel_launch(void* const* d_in, const int* in_sizes, int n_in,
                              void* d_out, int out_size, void* d_ws, size_t ws_size,
                              hipStream_t stream) {
  const float* x_a  = (const float*)d_in[0];
  const float* x_v  = (const float*)d_in[1];
  const float* x    = (const float*)d_in[2];
  const float* ln_g = (const float*)d_in[13];
  const float* ln_b = (const float*)d_in[14];

  char* ws = (char*)d_ws;
  size_t off = 0;
  auto alloc = [&](size_t bytes) -> void* {
    void* p = (void*)(ws + off);
    off = (off + bytes + 255) & ~(size_t)255;
    return p;
  };
  const size_t nel = (size_t)LDIM * BDIM * DDIM;  // 12,582,912
  unsigned short* qbf    = (unsigned short*)alloc(nel * 2);
  unsigned short* kbf    = (unsigned short*)alloc(nel * 2);
  unsigned short* ktbf   = (unsigned short*)alloc(nel * 2);
  float*          Sbuf   = (float*)alloc((size_t)LDIM * LDIM * 4);
  unsigned short* Pbuf   = (unsigned short*)alloc((size_t)LDIM * LDIM * 2);
  float*          stats  = (float*)alloc((size_t)LDIM * 2 * 4);
  unsigned short* ctx    = (unsigned short*)alloc(nel * 2);
  float*          ybuf   = (float*)alloc(nel * 4);
  unsigned short* wbf    = (unsigned short*)alloc((size_t)DDIM * DDIM * 2);
  float*          bnsums = (float*)alloc((size_t)2 * DDIM * 4);
  float*          bnss   = (float*)alloc((size_t)2 * DDIM * 4);

  // out = x (residual base), layout [L,B,D]
  (void)hipMemcpyAsync(d_out, (const void*)x, nel * 4, hipMemcpyDeviceToDevice, stream);
  k_cast_q<<<(int)(nel / 256), 256, 0, stream>>>(x, qbf);

  const float scl = 0.036084391824351614f;  // 1/sqrt(768)

  for (int br = 0; br < 2; ++br) {
    const float* xsrc  = br ? x_v : x_a;
    const float* W     = (const float*)d_in[br ? 8 : 3];
    const float* bias  = (const float*)d_in[br ? 9 : 4];
    const float* bg    = (const float*)d_in[br ? 10 : 5];
    const float* bb    = (const float*)d_in[br ? 11 : 6];
    const float* alpha = (const float*)d_in[br ? 12 : 7];

    k_cast_kv<<<(int)(nel / 256), 256, 0, stream>>>(xsrc, kbf, ktbf);
    k_cast_w<<<(DDIM * DDIM) / 256, 256, 0, stream>>>(W, wbf);

    for (int b = 0; b < BDIM; ++b) {
      const unsigned short* qb  = qbf  + (size_t)b * LDIM * DDIM;
      const unsigned short* kb  = kbf  + (size_t)b * LDIM * DDIM;
      const unsigned short* ktb = ktbf + (size_t)b * DDIM * LDIM;
      unsigned short*       cb  = ctx  + (size_t)b * LDIM * DDIM;

      // S = scale * Q K^T   (M=2048, N=2048, K=768) : 16x16 blocks of 128x128
      k_gemm<DDIM, LDIM, 0><<<256, 256, 0, stream>>>(qb, kb, (void*)Sbuf, nullptr, scl);
      k_rowstats<<<LDIM / 8, 256, 0, stream>>>(Sbuf, stats);
      k_softmax<<<(int)(((size_t)LDIM * LDIM) / 256), 256, 0, stream>>>(Sbuf, stats, Pbuf);
      // ctx = P KV          (M=2048, N=768, K=2048)
      k_gemm<LDIM, DDIM, 1><<<96, 256, 0, stream>>>(Pbuf, ktb, (void*)cb, nullptr, 1.0f);
    }
    // y = ctx W^T + bias    (M=16384, N=768, K=768)
    k_gemm<DDIM, DDIM, 2><<<768, 256, 0, stream>>>(ctx, wbf, (void*)ybuf, bias, 1.0f);

    k_zero<<<(2 * DDIM) / 256, 256, 0, stream>>>(bnsums);
    k_bnacc<<<NROWS / 64, 256, 0, stream>>>(ybuf, bnsums);
    k_bnfin<<<DDIM / 256, 256, 0, stream>>>(bnsums, bg, bb, bnss);
    k_apply<<<(int)(nel / 256), 256, 0, stream>>>(ybuf, bnss, alpha, (float*)d_out);
  }

  k_ln<<<NROWS, 256, 0, stream>>>((float*)d_out, ln_g, ln_b);
}
